// CrossMultiheadAttentionAug_7146825580821
// MI455X (gfx1250) — matrix operand order repaired and checked by reading, not compiled, not run on hardware
//
#include <hip/hip_runtime.h>

// MI455X / gfx1250: wave32, WMMA 16x16x32 f16 -> f32 accumulate,
// TDM (tensor_load_to_lds) double-buffered K/V staging in attention.

typedef __attribute__((ext_vector_type(16))) _Float16     v16h;
typedef __attribute__((ext_vector_type(8)))  float        v8f;
typedef __attribute__((ext_vector_type(4)))  unsigned int u32x4;
typedef __attribute__((ext_vector_type(8)))  int          i32x8;
typedef __attribute__((ext_vector_type(4)))  int          i32x4;

#define B_  4
#define T_  1024
#define S_  1024
#define E_  1024
#define H_  16
#define DH_ 64
#define HH_ 8

#if __has_builtin(__builtin_amdgcn_tensor_load_to_lds) && \
    __has_builtin(__builtin_amdgcn_s_wait_tensorcnt)
#define HAVE_TDM 1
#else
#define HAVE_TDM 0
#endif

// ---------------- WMMA fragment loaders (contiguous k = 16*(lane/16) + s, A and B alike) ------

// A fragment: 16x32 f16 (MxK). m = lane&15; lanes 0-15 hold K=0..15,
// lanes 16-31 hold K=16..31 (matches B fragment order).
static __device__ __forceinline__ v16h wmma_a_frag(const _Float16* tile,
                                                   int stride, int k0) {
  const int lane = threadIdx.x & 31;
  const int m    = lane & 15;
  const int hb   = (lane >> 4) << 4;
  const _Float16* row = tile + (size_t)m * stride + k0;
  v16h a;
#pragma unroll
  for (int j = 0; j < 4; ++j) {
    a[2 * j]         = row[hb + 2 * j];
    a[2 * j + 1]     = row[hb + 2 * j + 1];
    a[8 + 2 * j]     = row[8 + hb + 2 * j];
    a[8 + 2 * j + 1] = row[8 + hb + 2 * j + 1];
  }
  return a;
}

// B fragment: 32x16 f16 (KxN), tile row-major [K][stride].
// n = lane&15; lanes 0-15 hold K=0..15, lanes 16-31 hold K=16..31.
static __device__ __forceinline__ v16h wmma_b_frag(const _Float16* tile,
                                                   int stride, int n0) {
  const int lane = threadIdx.x & 31;
  const int n    = lane & 15;
  const int kb   = (lane >> 4) << 4;
  v16h b;
#pragma unroll
  for (int i = 0; i < 16; ++i)
    b[i] = tile[(size_t)(kb + i) * stride + n0 + n];
  return b;
}

// B fragment from transposed storage: tile is [N][stride] row-major,
// B[k][n] = tile[n][k]; per-lane reads are 16 contiguous f16.
static __device__ __forceinline__ v16h wmma_bT_frag(const _Float16* tile,
                                                    int stride, int n0, int k0) {
  const int lane = threadIdx.x & 31;
  const int n    = lane & 15;
  const int kb   = (lane >> 4) << 4;
  const _Float16* row = tile + (size_t)(n0 + n) * stride + k0 + kb;
  v16h b;
#pragma unroll
  for (int i = 0; i < 16; ++i) b[i] = row[i];
  return b;
}

// Row-max across the 16-lane half that owns a C-layout row.
// ds_swizzle immediates: group-of-32 mode, and_mask=0x1f, xor_mask=1/2/4/8.
#define SWZ_MAX(v, imm)                                                       \
  v = fmaxf(v, __int_as_float(                                                \
                   __builtin_amdgcn_ds_swizzle(__float_as_int(v), (imm))))
static __device__ __forceinline__ float half_rowmax(float v) {
  SWZ_MAX(v, 0x041f);   // xor 1
  SWZ_MAX(v, 0x081f);   // xor 2
  SWZ_MAX(v, 0x101f);   // xor 4
  SWZ_MAX(v, 0x201f);   // xor 8
  return v;
}

// ---------------- Tensor Data Mover: 64x64 f16 tile, row stride 64 ---------
#if HAVE_TDM
static __device__ __forceinline__ void tdm_load_tile_64x64(
    unsigned int lds_byte_off, const _Float16* gsrc) {
  unsigned long long ga = (unsigned long long)(uintptr_t)gsrc;
  u32x4 g0;
  g0[0] = 1u;                                   // count=1, user descriptor
  g0[1] = lds_byte_off;                         // lds_addr (bytes)
  g0[2] = (unsigned int)(ga & 0xffffffffu);     // global_addr[31:0]
  g0[3] = (unsigned int)((ga >> 32) & 0x01ffffffu) | (2u << 30);  // [56:32]|type=2

  // group1 bitfields assembled as 4 x u64 (see D# group1 layout)
  const unsigned long long dim0 = 64, dim1 = 64, td0 = 64, td1 = 64,
                           str0 = 64;
  unsigned long long q0 = (1ull << 16)                  // data_size = 2 bytes
                        | ((dim0 & 0xffffull) << 48);   // tensor_dim0[15:0]
  unsigned long long q1 = ((dim0 >> 16) & 0xffffull)    // tensor_dim0[31:16]
                        | (dim1 << 16)                  // tensor_dim1[31:0]
                        | (td0 << 48);                  // tile_dim0
  unsigned long long q2 = td1                           // tile_dim1
                        | ((str0 & 0xffffffffull) << 32);  // dim0_stride[31:0]
  unsigned long long q3 = ((str0 >> 32) & 0xffffull);   // dim0_stride[47:32]
  i32x8 g1;
  g1[0] = (int)(unsigned int)q0;  g1[1] = (int)(unsigned int)(q0 >> 32);
  g1[2] = (int)(unsigned int)q1;  g1[3] = (int)(unsigned int)(q1 >> 32);
  g1[4] = (int)(unsigned int)q2;  g1[5] = (int)(unsigned int)(q2 >> 32);
  g1[6] = (int)(unsigned int)q3;  g1[7] = (int)(unsigned int)(q3 >> 32);
  i32x4 z4 = {0, 0, 0, 0};
#if defined(__clang_major__) && __clang_major__ >= 23
  i32x8 z8 = {0, 0, 0, 0, 0, 0, 0, 0};
  __builtin_amdgcn_tensor_load_to_lds(g0, g1, z4, z4, z8, 0);
#else
  __builtin_amdgcn_tensor_load_to_lds(g0, g1, z4, z4, 0);
#endif
}
#endif

// ---------------- Projection GEMM: out = (A @ W^T + bias) * scale ----------
template <bool A_HALF, bool OUT_F32>
__global__ __launch_bounds__(256) void gemm_wmma(const void* __restrict__ Ain,
                                                 const float* __restrict__ W,
                                                 const float* __restrict__ bias,
                                                 void* __restrict__ Out,
                                                 float scale) {
  __shared__ _Float16 As[128 * 32];
  __shared__ _Float16 Bs[32 * 64];

  const int tid  = threadIdx.x;
  const int wave = tid >> 5;
  const int n0   = blockIdx.x * 64;
  const int m0   = blockIdx.y * 128;

  v8f acc[4] = {};

  for (int kc = 0; kc < E_ / 32; ++kc) {
#pragma unroll
    for (int it = 0; it < 16; ++it) {
      int idx = it * 256 + tid;
      int r = idx >> 5, c = idx & 31;
      size_t g = (size_t)(m0 + r) * E_ + kc * 32 + c;
      if (A_HALF) As[idx] = ((const _Float16*)Ain)[g];
      else        As[idx] = (_Float16)((const float*)Ain)[g];
      if (it == 0 && kc + 1 < E_ / 32) {
        if (A_HALF) __builtin_prefetch(((const _Float16*)Ain) + g + 32, 0, 0);
        else        __builtin_prefetch(((const float*)Ain) + g + 32, 0, 0);
      }
    }
#pragma unroll
    for (int it = 0; it < 8; ++it) {
      int idx = it * 256 + tid;
      int n = idx >> 5, k = idx & 31;
      size_t g = (size_t)(n0 + n) * E_ + kc * 32 + k;
      Bs[k * 64 + n] = (_Float16)W[g];
      if (it == 0 && kc + 1 < E_ / 32) __builtin_prefetch(W + g + 32, 0, 0);
    }
    __syncthreads();

    v16h a = wmma_a_frag(As + wave * 16 * 32, 32, 0);
#pragma unroll
    for (int j = 0; j < 4; ++j) {
      v16h b = wmma_b_frag(Bs, 64, 16 * j);
      acc[j] = __builtin_amdgcn_wmma_f32_16x16x32_f16(
          false, a, false, b, (short)0, acc[j], false, false);
    }
    __syncthreads();
  }

  const int lane = tid & 31;
  const int nlo  = lane & 15;
  const int half = lane >> 4;
#pragma unroll
  for (int j = 0; j < 4; ++j) {
#pragma unroll
    for (int r = 0; r < 8; ++r) {
      int m = m0 + wave * 16 + r + half * 8;
      int n = n0 + 16 * j + nlo;
      float v = (acc[j][r] + bias[n]) * scale;
      if (OUT_F32) {
        ((float*)Out)[(size_t)m * E_ + n] = v;
      } else {
        int bb = m >> 10, t = m & 1023;
        int h  = n >> 6,  d = n & 63;
        ((_Float16*)Out)[(((size_t)bb * H_ + h) * T_ + t) * DH_ + d] =
            (_Float16)v;
      }
    }
  }
}

// ---------------- Fused flash attention ------------------------------------
// Block = 8 waves over one (b,h,128 query rows); S walked in 64-wide chunks.
// K/V chunks DMA'd into double-buffered LDS by the TDM (wave 0 issues,
// s_wait_tensorcnt 2 retires the in-flight chunk, barrier releases waves).
// Row sums come from an extra WMMA against a ones tile (no lane shuffles);
// row max uses ds_swizzle XOR reductions across each 16-lane half.
__global__ __launch_bounds__(256) void flash_attn_wmma(
    const _Float16* __restrict__ Qh, const _Float16* __restrict__ Kh,
    const _Float16* __restrict__ Vh, const unsigned char* __restrict__ kpm,
    const unsigned char* __restrict__ lm, _Float16* __restrict__ O) {
  __shared__ _Float16 Ks[2][64 * DH_];
  __shared__ _Float16 Vs[2][64 * DH_];
  __shared__ _Float16 Ps[8][16 * 64];

  const int tid  = threadIdx.x;
  const int wave = tid >> 5;
  const int lane = tid & 31;
  const int nlo  = lane & 15;
  const int half = lane >> 4;

  const int b  = blockIdx.z;
  const int h  = blockIdx.y;
  const int t0 = blockIdx.x * 128 + wave * 16;

  const unsigned char* mask =
      (h < HH_) ? (kpm + (size_t)b * S_) : (lm + (size_t)b * S_);

  const _Float16* Qbase = Qh + (((size_t)b * H_ + h) * T_ + t0) * DH_;
  const _Float16* Kbase = Kh + (((size_t)b * H_ + h) * S_) * DH_;
  const _Float16* Vbase = Vh + (((size_t)b * H_ + h) * S_) * DH_;

  v16h qa[2];
  qa[0] = wmma_a_frag(Qbase, DH_, 0);
  qa[1] = wmma_a_frag(Qbase, DH_, 32);

  v16h ones;
#pragma unroll
  for (int i = 0; i < 16; ++i) ones[i] = (_Float16)1.0f;

  float mrun[8], lrun[8];
#pragma unroll
  for (int r = 0; r < 8; ++r) { mrun[r] = -1e30f; lrun[r] = 0.f; }
  v8f acc[4] = {};

  _Float16* Pw = Ps[wave];
  const int NC = S_ / 64;   // 16 chunks

#if HAVE_TDM
  const unsigned int ldsK[2] = {(unsigned int)(uintptr_t)&Ks[0][0],
                                (unsigned int)(uintptr_t)&Ks[1][0]};
  const unsigned int ldsV[2] = {(unsigned int)(uintptr_t)&Vs[0][0],
                                (unsigned int)(uintptr_t)&Vs[1][0]};
  if (wave == 0) {
    tdm_load_tile_64x64(ldsK[0], Kbase);
    tdm_load_tile_64x64(ldsV[0], Vbase);
  }
#endif

  for (int sc = 0; sc < NC; ++sc) {
    const int cur = sc & 1;
#if HAVE_TDM
    if (wave == 0) {
      if (sc + 1 < NC) {        // prefetch next chunk into the other buffer
        tdm_load_tile_64x64(ldsK[cur ^ 1], Kbase + (size_t)(sc + 1) * 64 * DH_);
        tdm_load_tile_64x64(ldsV[cur ^ 1], Vbase + (size_t)(sc + 1) * 64 * DH_);
        __builtin_amdgcn_s_wait_tensorcnt(2);   // current chunk retired
      } else {
        __builtin_amdgcn_s_wait_tensorcnt(0);
      }
    }
#else
    {   // cooperative fallback copy: 8 KB per tile, 2 x uint4 per thread
      const uint4* ksrc = (const uint4*)(Kbase + (size_t)sc * 64 * DH_);
      const uint4* vsrc = (const uint4*)(Vbase + (size_t)sc * 64 * DH_);
      uint4* kdst = (uint4*)&Ks[cur][0];
      uint4* vdst = (uint4*)&Vs[cur][0];
      kdst[tid]       = ksrc[tid];
      kdst[tid + 256] = ksrc[tid + 256];
      vdst[tid]       = vsrc[tid];
      vdst[tid + 256] = vsrc[tid + 256];
    }
#endif
    __syncthreads();

    const _Float16* Kc = Ks[cur];
    const _Float16* Vc = Vs[cur];

    // scores: four 16-wide s tiles, K-dim = DH = 2 x 32
    v8f sfrag[4];
#pragma unroll
    for (int c = 0; c < 4; ++c) {
      v8f z = {};
      v16h bk0 = wmma_bT_frag(Kc, DH_, 16 * c, 0);
      v16h bk1 = wmma_bT_frag(Kc, DH_, 16 * c, 32);
      z = __builtin_amdgcn_wmma_f32_16x16x32_f16(false, qa[0], false, bk0,
                                                 (short)0, z, false, false);
      z = __builtin_amdgcn_wmma_f32_16x16x32_f16(false, qa[1], false, bk1,
                                                 (short)0, z, false, false);
      if (mask[sc * 64 + 16 * c + nlo]) {
#pragma unroll
        for (int r = 0; r < 8; ++r) z[r] = -1e30f;
      }
      sfrag[c] = z;
    }

    // online softmax: rescale running state, emit P tile to LDS (A-layout src)
#pragma unroll
    for (int r = 0; r < 8; ++r) {
      float v = fmaxf(fmaxf(sfrag[0][r], sfrag[1][r]),
                      fmaxf(sfrag[2][r], sfrag[3][r]));
      v = half_rowmax(v);
      float mn    = fmaxf(mrun[r], v);
      float alpha = __expf(mrun[r] - mn);
      mrun[r] = mn;
      lrun[r] *= alpha;
#pragma unroll
      for (int j = 0; j < 4; ++j) acc[j][r] *= alpha;
      int m = r + half * 8;
#pragma unroll
      for (int c = 0; c < 4; ++c)
        Pw[m * 64 + 16 * c + nlo] = (_Float16)__expf(sfrag[c][r] - mn);
    }
    __syncthreads();

    v16h pa0 = wmma_a_frag(Pw, 64, 0);
    v16h pa1 = wmma_a_frag(Pw, 64, 32);

    // row sums of exp(P) via WMMA against ones: every column holds the sum
    v8f zsum = {};
    zsum = __builtin_amdgcn_wmma_f32_16x16x32_f16(false, pa0, false, ones,
                                                  (short)0, zsum, false, false);
    zsum = __builtin_amdgcn_wmma_f32_16x16x32_f16(false, pa1, false, ones,
                                                  (short)0, zsum, false, false);
#pragma unroll
    for (int r = 0; r < 8; ++r) lrun[r] += zsum[r];

    // acc += P (16x64) x V (64x64)
#pragma unroll
    for (int j = 0; j < 4; ++j) {
      v16h bv0 = wmma_b_frag(Vc, DH_, 16 * j);
      v16h bv1 = wmma_b_frag(Vc + 32 * DH_, DH_, 16 * j);
      acc[j] = __builtin_amdgcn_wmma_f32_16x16x32_f16(
          false, pa0, false, bv0, (short)0, acc[j], false, false);
      acc[j] = __builtin_amdgcn_wmma_f32_16x16x32_f16(
          false, pa1, false, bv1, (short)0, acc[j], false, false);
    }
    __syncthreads();   // all waves done with this buffer before TDM reuse
  }

  // normalize and emit O[b, t, h*64+d] (f16 feeds the output projection)
#pragma unroll
  for (int r = 0; r < 8; ++r) {
    float inv = 1.0f / lrun[r];
    int t = t0 + r + half * 8;
#pragma unroll
    for (int j = 0; j < 4; ++j) {
      int e = h * DH_ + 16 * j + nlo;
      O[((size_t)b * T_ + t) * E_ + e] = (_Float16)(acc[j][r] * inv);
    }
  }
}

// ---------------- host-side orchestration ----------------------------------
extern "C" void kernel_launch(void* const* d_in, const int* in_sizes, int n_in,
                              void* d_out, int out_size, void* d_ws,
                              size_t ws_size, hipStream_t stream) {
  (void)in_sizes; (void)n_in; (void)out_size; (void)ws_size;

  const float* query = (const float*)d_in[0];
  const float* key   = (const float*)d_in[1];
  const float* value = (const float*)d_in[2];
  const unsigned char* kpm = (const unsigned char*)d_in[3];
  const unsigned char* lm  = (const unsigned char*)d_in[4];
  const float* Wq = (const float*)d_in[5];
  const float* bq = (const float*)d_in[6];
  const float* Wk = (const float*)d_in[7];
  const float* bk = (const float*)d_in[8];
  const float* Wv = (const float*)d_in[9];
  const float* bv = (const float*)d_in[10];
  const float* Wo = (const float*)d_in[11];
  const float* bo = (const float*)d_in[12];

  const size_t headElems = (size_t)B_ * H_ * T_ * DH_;
  _Float16* Qh = (_Float16*)d_ws;
  _Float16* Kh = Qh + headElems;
  _Float16* Vh = Kh + headElems;
  _Float16* Oh = Vh + headElems;              // [B,T,E] f16

  dim3 blk(256);
  dim3 gproj(E_ / 64, (B_ * T_) / 128);

  gemm_wmma<false, false><<<gproj, blk, 0, stream>>>(query, Wq, bq, Qh, 0.125f);
  gemm_wmma<false, false><<<gproj, blk, 0, stream>>>(key,   Wk, bk, Kh, 1.0f);
  gemm_wmma<false, false><<<gproj, blk, 0, stream>>>(value, Wv, bv, Vh, 1.0f);

  dim3 gattn(T_ / 128, H_, B_);
  flash_attn_wmma<<<gattn, blk, 0, stream>>>(Qh, Kh, Vh, kpm, lm, Oh);

  gemm_wmma<true, true><<<gproj, blk, 0, stream>>>(Oh, Wo, bo, d_out, 1.0f);
}
